// FPN3DWithMultiHeadAttention_20392504721953
// MI455X (gfx1250) — matrix operand order repaired and checked by reading, not compiled, not run on hardware
//
#include <hip/hip_runtime.h>
#include <cstddef>

typedef __attribute__((ext_vector_type(16))) _Float16     v16h;
typedef __attribute__((ext_vector_type(8)))  float        v8f;
typedef __attribute__((ext_vector_type(4)))  _Float16     v4h;
typedef __attribute__((ext_vector_type(4)))  unsigned int uint4v;  // 16B, 16B-aligned
typedef _Float16 half_t;

#define N_TOK   4096
#define C_DIM   128
#define HEADS   8
#define HD      16
#define TAPS    27

#if __has_builtin(__builtin_amdgcn_global_load_async_to_lds_b128)
#define HAVE_ASYNC_LDS 1
#endif

// Exact pointee type the async-LDS builtins expect: vector-of-4-int.
typedef int int4v __attribute__((__vector_size__(16)));
typedef __attribute__((address_space(1))) int4v g_int4;   // global
typedef __attribute__((address_space(3))) int4v l_int4;   // LDS

// One 16-byte global->LDS stage. Async (no VGPR round trip, ASYNCcnt) when
// available; plain vector copy otherwise.
__device__ inline void stage_b128(void* lds, const void* g) {
#ifdef HAVE_ASYNC_LDS
  __builtin_amdgcn_global_load_async_to_lds_b128(
      (g_int4*)g, (l_int4*)lds, 0, 0);
#else
  *(uint4v*)lds = *(const uint4v*)g;
#endif
}

__device__ inline void stage_fence() {
#ifdef HAVE_ASYNC_LDS
#if __has_builtin(__builtin_amdgcn_s_wait_asynccnt)
  __builtin_amdgcn_s_wait_asynccnt(0);
#else
  asm volatile("s_wait_asynccnt 0" ::: "memory");
#endif
#endif
}

__device__ inline v8f zero8() {
  v8f z;
#pragma unroll
  for (int i = 0; i < 8; ++i) z[i] = 0.f;
  return z;
}

__device__ inline v8f wmma_f16(v16h a, v16h b, v8f c) {
  // D = A(16x32 f16) * B(32x16 f16) + C(16x16 f32)
  return __builtin_amdgcn_wmma_f32_16x16x32_f16(
      false, a, false, b, (short)0, c, false, false);
}

// A-fragment: A[16 rows][32 K] row-major, leading dim `ld` halfs.
// lane<16: row=lane, K {0..15}; lane>=16: row=lane-16, K {16..31}.
__device__ inline v16h load_a_frag(const half_t* base, int ld) {
  const int lane = threadIdx.x & 31;
  const half_t* p = base + (lane & 15) * ld + ((lane >> 4) << 4);
  v16h r;
#pragma unroll
  for (int i = 0; i < 16; ++i) r[i] = p[i];
  return r;
}

// B-fragment from TRANSPOSED storage: BT[16 N][32 K] row-major, ld halfs.
__device__ inline v16h load_bT_frag(const half_t* baseT, int ld) {
  const int lane = threadIdx.x & 31;
  const half_t* p = baseT + (lane & 15) * ld + ((lane >> 4) << 4);
  v16h r;
#pragma unroll
  for (int i = 0; i < 16; ++i) r[i] = p[i];
  return r;
}

// ---------------------------------------------------------------------------
// Weight repack (f32 -> f16)
// ---------------------------------------------------------------------------
__global__ void pack_lat_w(const float* __restrict__ w, half_t* __restrict__ out) {
  int i = blockIdx.x * 256 + threadIdx.x;                  // [27][128 co][64 ci]
  if (i >= TAPS * 128 * 64) return;
  int t = i / (128 * 64), r = i % (128 * 64);
  int co = r / 64, ci = r % 64;
  out[i] = (half_t)w[co * 64 * TAPS + ci * TAPS + t];
}

__global__ void pack_out_w(const float* __restrict__ w, half_t* __restrict__ out) {
  int i = blockIdx.x * 256 + threadIdx.x;                  // [27][128 o][128 ci]
  if (i >= TAPS * 128 * 128) return;
  int t = i / (128 * 128), r = i % (128 * 128);
  int o = r / 128, ci = r % 128;
  // ConvTranspose3d(k3,s1,p1) == conv with flipped, channel-transposed kernel
  out[i] = (half_t)w[ci * 128 * TAPS + o * TAPS + (TAPS - 1 - t)];
}

__global__ void pack_qkv_w(const float* __restrict__ qw, const float* __restrict__ kw,
                           const float* __restrict__ vw, half_t* __restrict__ out) {
  // dst layout [layer][3][o][k] == torch-native [out][in] per matrix (BT layout)
  int i = blockIdx.x * 256 + threadIdx.x;
  if (i >= 3 * 3 * 128 * 128) return;
  int l = i / (3 * 16384), r = i % (3 * 16384);
  int wsel = r / 16384, rr = r % 16384;
  const float* src = (wsel == 0) ? qw : (wsel == 1) ? kw : vw;
  out[i] = (half_t)src[l * 16384 + rr];
}

// lat f32 [c][n]  ->  xf16 [n][c]   (token-major f16 activations for QKV GEMM A)
__global__ void lat_to_xf16(const float* __restrict__ lat, half_t* __restrict__ xf) {
  int idx = blockIdx.x * 256 + threadIdx.x;                // 128*4096
  int c = idx >> 12;
  int n = idx & 4095;                                      // coalesced reads along n
  xf[(size_t)n * C_DIM + c] = (half_t)lat[(size_t)c * N_TOK + n];
}

// ---------------------------------------------------------------------------
// Conv3d (k=3, pad=1) as tap-wise implicit GEMM:
//   C[128 co][4096 vox] = sum_t  W_t[128][Cin] x X_shift_t[Cin][4096]
// ---------------------------------------------------------------------------
__global__ __launch_bounds__(256) void conv3d_tapgemm(
    const float* __restrict__ in,    // [Cin][4096] f32
    const half_t* __restrict__ w16,  // [27][128][Cin] f16
    const float* __restrict__ bias,  // [128]
    const float* __restrict__ pos,   // [128][4096] or nullptr
    float* __restrict__ out,         // [128][4096]
    int Cin)
{
  __shared__ __attribute__((aligned(16))) half_t a_lds[128 * 32];   // A  [co][ci]
  __shared__ __attribute__((aligned(16))) half_t b_lds[128 * 32];   // BT [vox][ci]

  const int tid  = threadIdx.x;
  const int wave = tid >> 5;
  const int lane = tid & 31;
  const int n0   = blockIdx.x * 128;
  const int mw   = wave >> 1;          // co base mw*32
  const int nw   = wave & 1;           // vox base nw*64

  v8f acc[2][4];
#pragma unroll
  for (int mi = 0; mi < 2; ++mi)
#pragma unroll
    for (int ni = 0; ni < 4; ++ni) acc[mi][ni] = zero8();

  for (int t = 0; t < TAPS; ++t) {
    const int dz = t / 9 - 1;
    const int dy = (t / 3) % 3 - 1;
    const int dx = t % 3 - 1;

    // Per-tap shifted-voxel addresses + 0/1 masks (independent of K chunk)
    int   snc[4];
    float okf[4];
#pragma unroll
    for (int j = 0; j < 4; ++j) {
      int vox = (j * 256 + tid) & 127;
      int n  = n0 + vox;
      int xx = (n & 15) + dx;
      int yy = ((n >> 4) & 15) + dy;
      int zz = (n >> 8) + dz;
      bool ok = ((unsigned)xx < 16u) && ((unsigned)yy < 16u) && ((unsigned)zz < 16u);
      snc[j] = ok ? ((zz << 8) + (yy << 4) + xx) : 0;   // clamped (safe) address
      okf[j] = ok ? 1.f : 0.f;
    }

    for (int kc = 0; kc < Cin; kc += 32) {
      __syncthreads();
      // Stage A: W_t[128 co][32 ci] -- async b128 copies
#pragma unroll
      for (int j = 0; j < 2; ++j) {
        int fi = j * 256 + tid;                       // 512 x uint4
        int co = fi >> 2, part = (fi & 3) << 3;
        stage_b128(a_lds + co * 32 + part,
                   w16 + (size_t)t * 128 * Cin + (size_t)co * Cin + kc + part);
      }
      // Stage BT: shifted input [128 vox][32 ci]; unconditional clamped loads
      // batched 4-wide, masked multiply, packed b64 store.
#pragma unroll
      for (int j = 0; j < 4; ++j) {
        int idx = j * 256 + tid;                      // 1024 quads
        int vox = idx & 127;
        int ci4 = (idx >> 7) << 2;
        float f[4];
#pragma unroll
        for (int u = 0; u < 4; ++u)
          f[u] = in[(size_t)(kc + ci4 + u) * N_TOK + snc[j]];
        v4h vals;
#pragma unroll
        for (int u = 0; u < 4; ++u)
          vals[u] = (half_t)(f[u] * okf[j]);
        *(v4h*)(b_lds + vox * 32 + ci4) = vals;
      }
      stage_fence();
      __syncthreads();

      v16h bf[4];
#pragma unroll
      for (int ni = 0; ni < 4; ++ni)
        bf[ni] = load_bT_frag(b_lds + (nw * 64 + ni * 16) * 32, 32);
#pragma unroll
      for (int mi = 0; mi < 2; ++mi) {
        v16h a = load_a_frag(a_lds + (mw * 32 + mi * 16) * 32, 32);
#pragma unroll
        for (int ni = 0; ni < 4; ++ni)
          acc[mi][ni] = wmma_f16(a, bf[ni], acc[mi][ni]);
      }
    }
  }

  // Epilogue: uniform branch hoisted so per-element loads batch cleanly
  const int mh = (lane >> 4) << 3;
  if (pos != nullptr) {
#pragma unroll
    for (int mi = 0; mi < 2; ++mi) {
#pragma unroll
      for (int ni = 0; ni < 4; ++ni) {
        int col = n0 + nw * 64 + ni * 16 + (lane & 15);
#pragma unroll
        for (int e = 0; e < 8; ++e) {
          int co = mw * 32 + mi * 16 + mh + e;
          out[(size_t)co * N_TOK + col] =
              acc[mi][ni][e] + bias[co] + pos[(size_t)co * N_TOK + col];
        }
      }
    }
  } else {
#pragma unroll
    for (int mi = 0; mi < 2; ++mi) {
#pragma unroll
      for (int ni = 0; ni < 4; ++ni) {
        int col = n0 + nw * 64 + ni * 16 + (lane & 15);
#pragma unroll
        for (int e = 0; e < 8; ++e) {
          int co = mw * 32 + mi * 16 + mh + e;
          out[(size_t)co * N_TOK + col] = acc[mi][ni][e] + bias[co];
        }
      }
    }
  }
}

// ---------------------------------------------------------------------------
// QKV projection: Y[tok][o] = Xf16[tok][128] * W^T + bias
// q,k -> [h][tok][16] ; v -> [h][16][tok] (d-major, for VT staging in attention)
// ---------------------------------------------------------------------------
__global__ __launch_bounds__(256) void qkv_proj(
    const half_t* __restrict__ xf,       // [4096][128] f16
    const half_t* __restrict__ wqkv,     // this layer: [3][128 o][128 k]
    const float* __restrict__ bq, const float* __restrict__ bk,
    const float* __restrict__ bv,
    half_t* __restrict__ q16, half_t* __restrict__ k16, half_t* __restrict__ v16)
{
  const int which = blockIdx.y;
  const half_t* w   = wqkv + (size_t)which * 128 * 128;
  const float* bias = (which == 0) ? bq : (which == 1) ? bk : bv;
  half_t* dst       = (which == 0) ? q16 : (which == 1) ? k16 : v16;

  __shared__ __attribute__((aligned(16))) half_t x_lds[128 * 32];   // A  [tok][k]
  __shared__ __attribute__((aligned(16))) half_t w_lds[128 * 32];   // BT [o][k]

  const int tid  = threadIdx.x;
  const int wave = tid >> 5;
  const int lane = tid & 31;
  const int t0   = blockIdx.x * 128;
  const int mw   = wave >> 1;
  const int nw   = wave & 1;

  v8f acc[2][4];
#pragma unroll
  for (int mi = 0; mi < 2; ++mi)
#pragma unroll
    for (int ni = 0; ni < 4; ++ni) acc[mi][ni] = zero8();

  for (int kc = 0; kc < 128; kc += 32) {
    __syncthreads();
#pragma unroll
    for (int j = 0; j < 2; ++j) {            // activations: async b128 copies
      int fi = j * 256 + tid;
      int tok = fi >> 2, part = (fi & 3) << 3;
      stage_b128(x_lds + tok * 32 + part,
                 xf + (size_t)(t0 + tok) * C_DIM + kc + part);
    }
#pragma unroll
    for (int j = 0; j < 2; ++j) {            // weights: async b128 copies
      int fi = j * 256 + tid;
      int o = fi >> 2, part = (fi & 3) << 3;
      stage_b128(w_lds + o * 32 + part,
                 w + (size_t)o * 128 + kc + part);
    }
    stage_fence();
    __syncthreads();

    v16h bf[4];
#pragma unroll
    for (int ni = 0; ni < 4; ++ni)
      bf[ni] = load_bT_frag(w_lds + (nw * 64 + ni * 16) * 32, 32);
#pragma unroll
    for (int mi = 0; mi < 2; ++mi) {
      v16h a = load_a_frag(x_lds + (mw * 32 + mi * 16) * 32, 32);
#pragma unroll
      for (int ni = 0; ni < 4; ++ni)
        acc[mi][ni] = wmma_f16(a, bf[ni], acc[mi][ni]);
    }
  }

  const int mh = (lane >> 4) << 3;
#pragma unroll
  for (int mi = 0; mi < 2; ++mi) {
#pragma unroll
    for (int ni = 0; ni < 4; ++ni) {
      int o = nw * 64 + ni * 16 + (lane & 15);
      int h = o >> 4, dd = o & 15;
#pragma unroll
      for (int e = 0; e < 8; ++e) {
        int tok = t0 + mw * 32 + mi * 16 + mh + e;
        float v = acc[mi][ni][e] + bias[o];
        size_t idx = (which == 2)
            ? (size_t)h * N_TOK * HD + (size_t)dd * N_TOK + tok   // v: d-major
            : (size_t)h * N_TOK * HD + (size_t)tok * HD + dd;     // q,k: tok-major
        dst[idx] = (half_t)v;
      }
    }
  }
}

// ---------------------------------------------------------------------------
// Flash attention: block = (head, 128 queries), wave = 16 queries.
// S = (Q/4) K^T (d=16 zero-padded to K=32), online softmax, O += P V.
// ---------------------------------------------------------------------------
__global__ __launch_bounds__(256) void flash_attn(
    const half_t* __restrict__ q16, const half_t* __restrict__ k16,
    const half_t* __restrict__ v16,                 // v16 is [h][d][tok]
    const float* __restrict__ latin, const float* __restrict__ gamma, int layer,
    float* __restrict__ latout, half_t* __restrict__ xfout)
{
  __shared__ __attribute__((aligned(16))) half_t k_lds[32 * HD + 32];  // [key][d] (+pad)
  __shared__ __attribute__((aligned(16))) half_t v_lds[HD * 32];       // VT [d][key]
  __shared__ __attribute__((aligned(16))) half_t p_lds[8][16 * 32];    // per-wave P

  const int tid  = threadIdx.x;
  const int wave = tid >> 5;
  const int lane = tid & 31;
  const int h     = blockIdx.y;
  const int qbase = blockIdx.x * 128 + wave * 16;
  const float g   = gamma[layer];
  const size_t hbase = (size_t)h * N_TOK * HD;

  // Q A-fragment, pre-scaled by 1/sqrt(16), K padded 16->32 with zeros
  v16h aq;
  {
    const half_t* qp = q16 + hbase + (size_t)(qbase + (lane & 15)) * HD;
#pragma unroll
    for (int i = 0; i < 16; ++i)
      aq[i] = (lane < 16) ? (half_t)((float)qp[i] * 0.25f) : (half_t)0.f;
  }

  float mrow[8], lrow[8];
  v8f oacc = zero8();
#pragma unroll
  for (int e = 0; e < 8; ++e) { mrow[e] = -3.0e38f; lrow[e] = 0.f; }

  for (int kc = 0; kc < N_TOK; kc += 32) {
    __syncthreads();
    // Stage K chunk [32 key][16 d] and VT chunk [16 d][32 key] (async b128)
    if (tid < 64) {
      stage_b128((half_t*)k_lds + tid * 8,
                 k16 + hbase + (size_t)kc * HD + tid * 8);
    } else if (tid < 128) {
      int fi = tid - 64;                   // 64 x uint4 == 16 rows x 32 halfs
      int dd = fi >> 2, part = (fi & 3) << 3;
      stage_b128(v_lds + dd * 32 + part,
                 v16 + hbase + (size_t)dd * N_TOK + kc + part);
    }
    stage_fence();
    __syncthreads();

    // K^T B-frags: contiguous per-lane 32B reads; padded K-half zeroed
    v16h bk0, bk1;
    {
      const int key = lane & 15;
      const int kb  = (lane >> 4) << 4;
      const half_t* p0 = k_lds + key * HD + kb;
      const half_t* p1 = k_lds + (16 + key) * HD + kb;
#pragma unroll
      for (int i = 0; i < 16; ++i) { bk0[i] = p0[i]; bk1[i] = p1[i]; }
      if (lane >= 16) {
#pragma unroll
        for (int i = 0; i < 16; ++i) { bk0[i] = (half_t)0.f; bk1[i] = (half_t)0.f; }
      }
    }
    v8f s0 = wmma_f16(aq, bk0, zero8());
    v8f s1 = wmma_f16(aq, bk1, zero8());

    // Online softmax (rows span 16 contiguous lanes)
    float p0v[8], p1v[8];
#pragma unroll
    for (int e = 0; e < 8; ++e) {
      float t = fmaxf(s0[e], s1[e]);
#pragma unroll
      for (int mask = 1; mask < 16; mask <<= 1)
        t = fmaxf(t, __shfl_xor(t, mask, 32));
      float nm = fmaxf(mrow[e], t);
      float alpha = __expf(mrow[e] - nm);
      p0v[e] = __expf(s0[e] - nm);
      p1v[e] = __expf(s1[e] - nm);
      float rs = p0v[e] + p1v[e];
#pragma unroll
      for (int mask = 1; mask < 16; mask <<= 1)
        rs += __shfl_xor(rs, mask, 32);
      lrow[e] = lrow[e] * alpha + rs;
      mrow[e] = nm;
      oacc[e] *= alpha;
    }

    // Spill P (C-layout) to per-wave LDS, re-read as A-fragment
    {
      half_t* pw = p_lds[wave];
      const int n = lane & 15, mh = (lane >> 4) << 3;
#pragma unroll
      for (int e = 0; e < 8; ++e) {
        pw[(mh + e) * 32 + n]      = (half_t)p0v[e];
        pw[(mh + e) * 32 + 16 + n] = (half_t)p1v[e];
      }
    }
    asm volatile("s_wait_dscnt 0" ::: "memory");   // wave-local LDS RAW fence

    v16h ap = load_a_frag(p_lds[wave], 32);
    v16h bv = load_bT_frag(v_lds, 32);             // B[key][d] via VT[d][key]
    oacc = wmma_f16(ap, bv, oacc);
  }

  // Epilogue: normalize, gamma*o + residual; also token-major f16 copy
  {
    const int dd = lane & 15, mh = (lane >> 4) << 3;
#pragma unroll
    for (int e = 0; e < 8; ++e) {
      int tok = qbase + mh + e;
      int c = h * HD + dd;
      float v = g * (oacc[e] / lrow[e]) + latin[(size_t)c * N_TOK + tok];
      latout[(size_t)c * N_TOK + tok] = v;
      xfout[(size_t)tok * C_DIM + c]  = (half_t)v;
    }
  }
}

// ---------------------------------------------------------------------------
// Host-side orchestration
// ---------------------------------------------------------------------------
extern "C" void kernel_launch(void* const* d_in, const int* in_sizes, int n_in,
                              void* d_out, int out_size, void* d_ws, size_t ws_size,
                              hipStream_t stream) {
  const float* x         = (const float*)d_in[0];
  const float* lateral_w = (const float*)d_in[1];
  const float* lateral_b = (const float*)d_in[2];
  const float* pos       = (const float*)d_in[3];
  const float* qw        = (const float*)d_in[4];
  const float* qb        = (const float*)d_in[5];
  const float* kw        = (const float*)d_in[6];
  const float* kb        = (const float*)d_in[7];
  const float* vw        = (const float*)d_in[8];
  const float* vb        = (const float*)d_in[9];
  const float* gamma     = (const float*)d_in[10];
  const float* out_w     = (const float*)d_in[11];
  const float* out_b     = (const float*)d_in[12];

  char* ws = (char*)d_ws;
  float*  latA   = (float*)(ws + 0);              // 2 MB
  float*  latB   = (float*)(ws + 2097152);        // 2 MB
  half_t* q16    = (half_t*)(ws + 4194304);       // 1 MB
  half_t* k16    = (half_t*)(ws + 5242880);       // 1 MB
  half_t* v16    = (half_t*)(ws + 6291456);       // 1 MB
  half_t* wl16   = (half_t*)(ws + 7340032);       // 432 KB
  half_t* wo16   = (half_t*)(ws + 7782400);       // 864 KB
  half_t* wqkv16 = (half_t*)(ws + 8667136);       // 288 KB
  half_t* xf16   = (half_t*)(ws + 8962048);       // 1 MB

  // 1) repack weights
  pack_lat_w<<<(TAPS * 128 * 64 + 255) / 256, 256, 0, stream>>>(lateral_w, wl16);
  pack_out_w<<<(TAPS * 128 * 128 + 255) / 256, 256, 0, stream>>>(out_w, wo16);
  pack_qkv_w<<<(3 * 3 * 128 * 128 + 255) / 256, 256, 0, stream>>>(qw, kw, vw, wqkv16);

  // 2) lateral conv + bias + pos -> latA ; seed token-major f16 copy
  conv3d_tapgemm<<<32, 256, 0, stream>>>(x, wl16, lateral_b, pos, latA, 64);
  lat_to_xf16<<<(C_DIM * N_TOK + 255) / 256, 256, 0, stream>>>(latA, xf16);

  // 3) attention stack
  float* cur = latA;
  float* nxt = latB;
  for (int l = 0; l < 3; ++l) {
    qkv_proj<<<dim3(32, 3), 256, 0, stream>>>(xf16, wqkv16 + (size_t)l * 3 * 128 * 128,
                                              qb + l * 128, kb + l * 128, vb + l * 128,
                                              q16, k16, v16);
    flash_attn<<<dim3(32, HEADS), 256, 0, stream>>>(q16, k16, v16, cur, gamma, l,
                                                    nxt, xf16);
    float* t = cur; cur = nxt; nxt = t;
  }

  // 4) output transposed conv (as flipped conv) -> d_out
  conv3d_tapgemm<<<32, 256, 0, stream>>>(cur, wo16, out_b, nullptr, (float*)d_out, 128);
}
